// HierarchyAwareGNN_48369921687559
// MI455X (gfx1250) — matrix operand order repaired and checked by reading, not compiled, not run on hardware
//
#include <hip/hip_runtime.h>
#include <math.h>

#define NNODES 16384
#define NEDGES 262144
#define FINDIM 256
#define NHEAD  4
#define CHDIM  128
#define DDIM   512

typedef __bf16 bf16;
typedef __attribute__((ext_vector_type(8)))  __bf16 bf16x8;
typedef __attribute__((ext_vector_type(16))) __bf16 bf16x16;
typedef __attribute__((ext_vector_type(8)))  float  f32x8;

// ---------- helpers ----------
__device__ __forceinline__ unsigned short f2bf(float x) {
  unsigned u = __float_as_uint(x);
  u += 0x7fffu + ((u >> 16) & 1u);          // round-to-nearest-even
  return (unsigned short)(u >> 16);
}

// ---------- elementwise kernels ----------
__global__ void fill_u32_k(unsigned* p, unsigned v, int n) {
  int i = blockIdx.x * blockDim.x + threadIdx.x;
  if (i < n) p[i] = v;
}
__global__ void conv_bf16_k(const float* __restrict__ in, unsigned short* __restrict__ out, int n) {
  int i = blockIdx.x * blockDim.x + threadIdx.x;
  if (i < n) out[i] = f2bf(in[i]);
}
// W [K][Nc] f32 -> Wt [Nc][K] bf16
__global__ void transp_bf16_k(const float* __restrict__ W, unsigned short* __restrict__ Wt, int K, int Nc) {
  int i = blockIdx.x * blockDim.x + threadIdx.x;
  if (i < K * Nc) {
    int k = i / Nc, n = i - k * Nc;
    Wt[(size_t)n * K + k] = f2bf(W[i]);
  }
}
// x = elu(x*g*inv + be)   (GAT bias already folded into aggregation)
__global__ void bn_elu_k(float* __restrict__ x, const float* __restrict__ g,
                         const float* __restrict__ be, float inv, int n, int Dc) {
  int i = blockIdx.x * blockDim.x + threadIdx.x;
  if (i < n) {
    int c = i % Dc;
    float v = x[i] * g[c] * inv + be[c];
    x[i] = v > 0.f ? v : (__expf(v) - 1.f);
  }
}
__global__ void elu_k(float* __restrict__ x, int n) {
  int i = blockIdx.x * blockDim.x + threadIdx.x;
  if (i < n) { float v = x[i]; x[i] = v > 0.f ? v : (__expf(v) - 1.f); }
}
// out[n] = z[n,:] . w + b
__global__ void dot_out_k(const float* __restrict__ z, const float* __restrict__ w,
                          const float* __restrict__ b, float* __restrict__ out, int n, int K) {
  int i = blockIdx.x * blockDim.x + threadIdx.x;
  if (i < n) {
    float s = b[0];
    for (int c = 0; c < K; ++c) s += z[(size_t)i * K + c] * w[c];
    out[i] = s;
  }
}

// ---------- WMMA GEMM: C[M,Nc] = A[M,K](bf16) * Bt[Nc,K]^T(bf16) (+bias +resid) ----------
// wave -> 16x64 tile (4 accumulators); block = 8 waves -> 128 rows x 64 cols.
// Software-pipelined: double-buffered K steps (2x 16x16x32 per iteration).
__global__ __launch_bounds__(256) void gemm_wmma_k(
    const bf16* __restrict__ A, const bf16* __restrict__ Bt,
    const float* __restrict__ bias, const float* __restrict__ resid,
    float* __restrict__ C, int M, int K, int Nc) {
  const int lane = threadIdx.x & 31;
  const int wave = threadIdx.x >> 5;
  const int m0 = (blockIdx.x * 8 + wave) * 16;
  const int n0 = blockIdx.y * 64;
  if (m0 >= M) return;
  const int cl = lane & 15;
  const int kA = (lane >> 4) * 8;    // A 16x32 bf16 layout: lane-half selects K 0-7 / 8-15 (+16)
  const int kB = (lane >> 4) * 8;    // B 32x16 bf16 layout: lane-half selects K 0-7 / 8-15 (+16)
  const bf16* Ap = A + (size_t)(m0 + cl) * K + kA;
  const bf16* Bp[4];
#pragma unroll
  for (int j = 0; j < 4; ++j) Bp[j] = Bt + (size_t)(n0 + j * 16 + cl) * K + kB;

  auto lda = [&](int k0, bf16x16& a) {
    bf16x8 lo = *(const bf16x8*)(Ap + k0);
    bf16x8 hi = *(const bf16x8*)(Ap + k0 + 16);
#pragma unroll
    for (int t = 0; t < 8; ++t) { a[t] = lo[t]; a[t + 8] = hi[t]; }
  };
  auto ldb = [&](const bf16* p, int k0, bf16x16& b) {
    bf16x8 lo = *(const bf16x8*)(p + k0);
    bf16x8 hi = *(const bf16x8*)(p + k0 + 16);
#pragma unroll
    for (int t = 0; t < 8; ++t) { b[t] = lo[t]; b[t + 8] = hi[t]; }
  };

  f32x8 acc[4] = {};
  bf16x16 a0, a1, b0[4], b1[4];
  lda(0, a0);
#pragma unroll
  for (int j = 0; j < 4; ++j) ldb(Bp[j], 0, b0[j]);

  for (int k0 = 0; k0 < K; k0 += 64) {
    __builtin_prefetch(Ap + k0 + 64, 0, 0);        // global_prefetch_b8
    // stage 2 loads (always valid: K % 64 == 0)
    lda(k0 + 32, a1);
#pragma unroll
    for (int j = 0; j < 4; ++j) ldb(Bp[j], k0 + 32, b1[j]);
    // stage 1 compute
#pragma unroll
    for (int j = 0; j < 4; ++j)
      acc[j] = __builtin_amdgcn_wmma_f32_16x16x32_bf16(
          false, a0, false, b0[j], (short)0, acc[j], false, false);
    // next iteration stage 1 loads
    if (k0 + 64 < K) {
      lda(k0 + 64, a0);
#pragma unroll
      for (int j = 0; j < 4; ++j) ldb(Bp[j], k0 + 64, b0[j]);
    }
    // stage 2 compute
#pragma unroll
    for (int j = 0; j < 4; ++j)
      acc[j] = __builtin_amdgcn_wmma_f32_16x16x32_bf16(
          false, a1, false, b1[j], (short)0, acc[j], false, false);
  }

  // D 16x16 f32: VGPR r -> M = r (lanes 0-15) / 8+r (lanes 16-31); N = lane&15
  const int rb = (lane >> 4) * 8;
#pragma unroll
  for (int j = 0; j < 4; ++j) {
#pragma unroll
    for (int r = 0; r < 8; ++r) {
      int row = m0 + rb + r;
      int col = n0 + j * 16 + cl;
      float v = acc[j][r];
      if (bias)  v += bias[col];
      if (resid) v += resid[(size_t)row * Nc + col];
      C[(size_t)row * Nc + col] = v;
    }
  }
}

// ---------- CSR build (done once per launch, reused by all 3 GAT layers) ----------
__global__ void hist_k(const int* __restrict__ dst, unsigned* __restrict__ deg, int EA, int E) {
  int e = blockIdx.x * blockDim.x + threadIdx.x;
  if (e >= EA) return;
  int d = (e < E) ? dst[e] : (e - E);
  atomicAdd(&deg[d], 1u);
}
// 2-level exclusive scan over N elements (blocks of 256)
__global__ void scan_block_k(const unsigned* __restrict__ in, unsigned* __restrict__ out,
                             unsigned* __restrict__ bsum, int n) {
  __shared__ unsigned tmp[256];
  int i = blockIdx.x * 256 + threadIdx.x;
  unsigned v = (i < n) ? in[i] : 0u;
  tmp[threadIdx.x] = v;
  __syncthreads();
  for (int off = 1; off < 256; off <<= 1) {
    unsigned t = (threadIdx.x >= off) ? tmp[threadIdx.x - off] : 0u;
    __syncthreads();
    tmp[threadIdx.x] += t;
    __syncthreads();
  }
  if (i < n) out[i] = tmp[threadIdx.x] - v;        // exclusive
  if (threadIdx.x == 255) bsum[blockIdx.x] = tmp[255];
}
__global__ void scan_sums_k(unsigned* bsum, int nb) {
  if (blockIdx.x == 0 && threadIdx.x == 0) {
    unsigned acc = 0;
    for (int i = 0; i < nb; ++i) { unsigned v = bsum[i]; bsum[i] = acc; acc += v; }
  }
}
__global__ void scan_add_k(unsigned* out, const unsigned* __restrict__ bsum, int n) {
  int i = blockIdx.x * 256 + threadIdx.x;
  if (i < n) out[i] += bsum[blockIdx.x];
}
__global__ void scatter_k(const int* __restrict__ dst, unsigned* __restrict__ cursor,
                          const unsigned* __restrict__ base, unsigned* __restrict__ elist,
                          int EA, int E) {
  int e = blockIdx.x * blockDim.x + threadIdx.x;
  if (e >= EA) return;
  int d = (e < E) ? dst[e] : (e - E);
  unsigned pos = base[d] + atomicAdd(&cursor[d], 1u);
  elist[pos] = (unsigned)e;
}
// per-node insertion sort of edge lists -> deterministic summation order
__global__ void sortlist_k(const unsigned* __restrict__ base, const unsigned* __restrict__ deg,
                           unsigned* __restrict__ elist, int Nn) {
  int n = blockIdx.x * blockDim.x + threadIdx.x;
  if (n >= Nn) return;
  unsigned b = base[n], dgr = deg[n];
  for (unsigned i = 1; i < dgr; ++i) {
    unsigned key = elist[b + i];
    int j = (int)i - 1;
    while (j >= 0 && elist[b + j] > key) { elist[b + j + 1] = elist[b + j]; --j; }
    elist[b + j + 1] = key;
  }
}

// ---------- GAT edge phase (atomic-free) ----------
// one wave per (node, head): ssrc/sdst = xw[n,h,:] . a_s/a_d
__global__ __launch_bounds__(256) void node_scores_k(
    const float* __restrict__ xw, const float* __restrict__ a_s, const float* __restrict__ a_d,
    float* __restrict__ ssrc, float* __restrict__ sdst, int Nn, int Hh, int Cc) {
  int task = blockIdx.x * 8 + (threadIdx.x >> 5);
  int lane = threadIdx.x & 31;
  if (task >= Nn * Hh) return;
  int n = task / Hh, h = task - n * Hh;
  const float* v = xw + (size_t)n * Hh * Cc + (size_t)h * Cc;
  float s1 = 0.f, s2 = 0.f;
  for (int c = lane; c < Cc; c += 32) {
    float x = v[c];
    s1 += x * a_s[h * Cc + c];
    s2 += x * a_d[h * Cc + c];
  }
  for (int off = 16; off; off >>= 1) {
    s1 += __shfl_xor(s1, off, 32);
    s2 += __shfl_xor(s2, off, 32);
  }
  if (lane == 0) { ssrc[task] = s1; sdst[task] = s2; }
}
// one wave per (dst, head): segment softmax over incoming edges, writes alpha[e*H+h]
__global__ __launch_bounds__(256) void seg_softmax_k(
    const int* __restrict__ src, const float* __restrict__ ssrc, const float* __restrict__ sdst,
    const unsigned* __restrict__ base, const unsigned* __restrict__ deg,
    const unsigned* __restrict__ elist, float* __restrict__ alpha, int Nn, int Hh, int E) {
  int task = blockIdx.x * 8 + (threadIdx.x >> 5);
  int lane = threadIdx.x & 31;
  if (task >= Nn * Hh) return;
  int d = task / Hh, h = task - d * Hh;
  unsigned b = base[d], dgr = deg[d];
  float sd = sdst[d * Hh + h];
  float mx = -3.4e38f;
  for (unsigned i = lane; i < dgr; i += 32) {
    unsigned e = elist[b + i];
    int s = (e < (unsigned)E) ? src[e] : (int)(e - E);
    float sc = ssrc[s * Hh + h] + sd;
    sc = (sc > 0.f) ? sc : 0.2f * sc;           // leaky_relu(0.2)
    mx = fmaxf(mx, sc);
  }
  for (int off = 16; off; off >>= 1) mx = fmaxf(mx, __shfl_xor(mx, off, 32));
  float sum = 0.f;
  for (unsigned i = lane; i < dgr; i += 32) {
    unsigned e = elist[b + i];
    int s = (e < (unsigned)E) ? src[e] : (int)(e - E);
    float sc = ssrc[s * Hh + h] + sd;
    sc = (sc > 0.f) ? sc : 0.2f * sc;
    float pe = __expf(sc - mx);
    alpha[e * Hh + h] = pe;
    sum += pe;
  }
  for (int off = 16; off; off >>= 1) sum += __shfl_xor(sum, off, 32);
  float rs = 1.f / sum;                          // deg >= 1 (self-loop)
  for (unsigned i = lane; i < dgr; i += 32) {
    unsigned e = elist[b + i];
    alpha[e * Hh + h] *= rs;
  }
}
// one wave per (dst, 128-float chunk): out[d,c] = bias[c] + sum_e alpha * xw[src,c]
__global__ __launch_bounds__(256) void seg_agg_k(
    const int* __restrict__ src, const float* __restrict__ xw, const float* __restrict__ alpha,
    const unsigned* __restrict__ base, const unsigned* __restrict__ deg,
    const unsigned* __restrict__ elist, const float* __restrict__ bias,
    float* __restrict__ outp, int Nn, int Hh, int Cc, int Dd, int E) {
  int wavesPerNode = Dd >> 7;                     // 128 floats per wave (32 lanes x float4)
  int task = blockIdx.x * 8 + (threadIdx.x >> 5);
  int lane = threadIdx.x & 31;
  if (task >= Nn * wavesPerNode) return;
  int d = task / wavesPerNode;
  int c0 = (task - d * wavesPerNode) * 128 + lane * 4;
  int h = c0 / Cc;
  unsigned b = base[d], dgr = deg[d];
  float4 acc = {0.f, 0.f, 0.f, 0.f};
  for (unsigned i = 0; i < dgr; ++i) {
    unsigned e = elist[b + i];
    int s = (e < (unsigned)E) ? src[e] : (int)(e - E);
    float a = alpha[e * Hh + h];
    const float4 v = *(const float4*)(xw + (size_t)s * Dd + c0);
    acc.x += v.x * a; acc.y += v.y * a; acc.z += v.z * a; acc.w += v.w * a;
  }
  float4 r;
  r.x = acc.x + bias[c0 + 0];
  r.y = acc.y + bias[c0 + 1];
  r.z = acc.z + bias[c0 + 2];
  r.w = acc.w + bias[c0 + 3];
  *(float4*)(outp + (size_t)d * Dd + c0) = r;
}

// ---------- host ----------
extern "C" void kernel_launch(void* const* d_in, const int* in_sizes, int n_in,
                              void* d_out, int out_size, void* d_ws, size_t ws_size,
                              hipStream_t stream) {
  const float* x    = (const float*)d_in[0];
  const int*   eidx = (const int*)  d_in[1];
  const float* W1   = (const float*)d_in[2];
  const float* as1  = (const float*)d_in[3];
  const float* ad1  = (const float*)d_in[4];
  const float* b1   = (const float*)d_in[5];
  const float* g1   = (const float*)d_in[6];
  const float* be1  = (const float*)d_in[7];
  const float* W2   = (const float*)d_in[8];
  const float* as2  = (const float*)d_in[9];
  const float* ad2  = (const float*)d_in[10];
  const float* b2   = (const float*)d_in[11];
  const float* g2   = (const float*)d_in[12];
  const float* be2  = (const float*)d_in[13];
  const float* W3   = (const float*)d_in[14];
  const float* as3  = (const float*)d_in[15];
  const float* ad3  = (const float*)d_in[16];
  const float* b3   = (const float*)d_in[17];
  const float* Wv   = (const float*)d_in[18];
  const float* bv   = (const float*)d_in[19];
  const float* Wo   = (const float*)d_in[20];
  const float* bo   = (const float*)d_in[21];
  const float* fw1  = (const float*)d_in[22];
  const float* fb1  = (const float*)d_in[23];
  const float* fw2  = (const float*)d_in[24];
  const float* fb2  = (const float*)d_in[25];
  float* out = (float*)d_out;

  const int EA = NEDGES + NNODES;
  const int* esrc = eidx;
  const int* edst = eidx + NEDGES;

  // workspace carve-up
  char* ws = (char*)d_ws;
  size_t off = 0;
  auto take = [&](size_t bytes) -> void* {
    void* r = ws + off;
    off += (bytes + 255) & ~(size_t)255;
    return r;
  };
  float*          bufA   = (float*)take((size_t)NNODES * DDIM * 4);
  float*          bufB   = (float*)take((size_t)NNODES * DDIM * 4);
  float*          bufC   = (float*)take((size_t)NNODES * CHDIM * 4);
  unsigned short* Xb     = (unsigned short*)take((size_t)NNODES * DDIM * 2);
  unsigned short* Wt     = (unsigned short*)take((size_t)DDIM * DDIM * 2);
  float*          ssrc   = (float*)take((size_t)NNODES * NHEAD * 4);
  float*          sdst   = (float*)take((size_t)NNODES * NHEAD * 4);
  unsigned*       deg    = (unsigned*)take((size_t)NNODES * 4);
  unsigned*       basep  = (unsigned*)take((size_t)NNODES * 4);
  unsigned*       cursor = (unsigned*)take((size_t)NNODES * 4);
  unsigned*       bsum   = (unsigned*)take((size_t)(NNODES / 256) * 4);
  unsigned*       elist  = (unsigned*)take((size_t)EA * 4);
  float*          alpha  = (float*)take((size_t)EA * NHEAD * 4);
  if (off > ws_size) return;  // not enough scratch

  auto cdiv = [](long long a, long long b) { return (int)((a + b - 1) / b); };
  const float inv = 0.99999500003749983f;  // 1/sqrt(1+1e-5)
  const int NB = NNODES / 256;             // 64 scan blocks

  // ---- CSR build (once; graph shared by all layers) ----
  fill_u32_k<<<cdiv(NNODES, 256), 256, 0, stream>>>(deg, 0u, NNODES);
  hist_k<<<cdiv(EA, 256), 256, 0, stream>>>(edst, deg, EA, NEDGES);
  scan_block_k<<<NB, 256, 0, stream>>>(deg, basep, bsum, NNODES);
  scan_sums_k<<<1, 32, 0, stream>>>(bsum, NB);
  scan_add_k<<<NB, 256, 0, stream>>>(basep, bsum, NNODES);
  fill_u32_k<<<cdiv(NNODES, 256), 256, 0, stream>>>(cursor, 0u, NNODES);
  scatter_k<<<cdiv(EA, 256), 256, 0, stream>>>(edst, cursor, basep, elist, EA, NEDGES);
  sortlist_k<<<cdiv(NNODES, 256), 256, 0, stream>>>(basep, deg, elist, NNODES);

  auto run_gemm = [&](const float* hin, int K, const float* W, int Nc,
                      const float* bias, const float* resid, float* outp) {
    conv_bf16_k<<<cdiv((long long)NNODES * K, 256), 256, 0, stream>>>(hin, Xb, NNODES * K);
    transp_bf16_k<<<cdiv((long long)K * Nc, 256), 256, 0, stream>>>(W, Wt, K, Nc);
    dim3 g(NNODES / 128, Nc / 64);
    gemm_wmma_k<<<g, 256, 0, stream>>>((const bf16*)Xb, (const bf16*)Wt,
                                       bias, resid, outp, NNODES, K, Nc);
  };
  auto run_edges = [&](const float* xw, const float* a_s, const float* a_d,
                       const float* bias, int Hh, int Dd, float* agg) {
    int Cc = Dd / Hh;
    node_scores_k<<<cdiv((long long)NNODES * Hh, 8), 256, 0, stream>>>(xw, a_s, a_d, ssrc, sdst, NNODES, Hh, Cc);
    seg_softmax_k<<<cdiv((long long)NNODES * Hh, 8), 256, 0, stream>>>(esrc, ssrc, sdst, basep, deg, elist, alpha, NNODES, Hh, NEDGES);
    seg_agg_k<<<cdiv((long long)NNODES * (Dd >> 7), 8), 256, 0, stream>>>(esrc, xw, alpha, basep, deg, elist, bias, agg, NNODES, Hh, Cc, Dd, NEDGES);
  };

  // ---- GAT layer 1: x[N,256] -> h1[N,512] ----
  run_gemm(x, FINDIM, W1, DDIM, nullptr, nullptr, bufA);            // xw1
  run_edges(bufA, as1, ad1, b1, NHEAD, DDIM, bufB);                 // agg+b1
  bn_elu_k<<<cdiv((long long)NNODES * DDIM, 256), 256, 0, stream>>>(bufB, g1, be1, inv, NNODES * DDIM, DDIM);

  // ---- GAT layer 2: h1[N,512] -> h2[N,512] ----
  run_gemm(bufB, DDIM, W2, DDIM, nullptr, nullptr, bufA);           // xw2
  run_edges(bufA, as2, ad2, b2, NHEAD, DDIM, bufB);
  bn_elu_k<<<cdiv((long long)NNODES * DDIM, 256), 256, 0, stream>>>(bufB, g2, be2, inv, NNODES * DDIM, DDIM);

  // ---- GAT layer 3: h2[N,512] -> h3[N,128] (H=1, mean over 1 head == identity) ----
  run_gemm(bufB, DDIM, W3, CHDIM, nullptr, nullptr, bufA);          // xw3
  run_edges(bufA, as3, ad3, b3, 1, CHDIM, bufC);                    // h3 = agg+b3

  // ---- MHA (seq_len==1): h = h3 + (h3@Wv+bv)@Wo + bo ----
  run_gemm(bufC, CHDIM, Wv, CHDIM, bv, nullptr, bufA);              // t
  run_gemm(bufA, CHDIM, Wo, CHDIM, bo, bufC, bufB);                 // h

  // ---- MLP head: out = elu(h@fw1+fb1) @ fw2 + fb2 ----
  run_gemm(bufB, CHDIM, fw1, CHDIM / 2, fb1, nullptr, bufA);        // z[N,64]
  elu_k<<<cdiv((long long)NNODES * (CHDIM / 2), 256), 256, 0, stream>>>(bufA, NNODES * (CHDIM / 2));
  dot_out_k<<<cdiv(NNODES, 256), 256, 0, stream>>>(bufA, fw2, fb2, out, NNODES, CHDIM / 2);
}
